// Mamba2RSEBlock_12077448036717
// MI455X (gfx1250) — matrix operand order repaired and checked by reading, not compiled, not run on hardware
//
#include <hip/hip_runtime.h>
#include <cstdint>
#include <cstddef>

// ---------------------------------------------------------------------------
// Mamba2 RSE block for MI455X (gfx1250), wave32 + bf16 WMMA pipeline.
//
//   0) f32->bf16 conversions / transposes of x and the weight matrices
//   1) zxbcdt = x @ in_proj_w^T                (bf16 WMMA, f32 accum)
//   2) dt = softplus(...), depthwise conv + SiLU (VALU)
//   3) lora: t1 = tanh(x@w1); theta/lambda -> log_z  (bf16 WMMA + VALU epilogue)
//   4) chunked complex scan via CB-trick: G = Re(P Q^T) (masked),
//      Y = G@Xd + P@(scaled state), state' = decay*state + Xd^T@Q'  (WMMA in LDS,
//      log_z staged via GLOBAL_LOAD_ASYNC_TO_LDS_B128 / ASYNCcnt)
//   5) gating + RMSNorm (VALU)   6) out = g @ out_proj_w^T (bf16 WMMA)
// ---------------------------------------------------------------------------

typedef __attribute__((ext_vector_type(16))) __bf16 v16bf;
typedef __attribute__((ext_vector_type(8)))  __bf16 v8bf;
typedef __attribute__((ext_vector_type(8)))  float  v8f;

#define DEVINL __device__ __forceinline__

DEVINL v16bf frag_cat(v8bf lo, v8bf hi) {
  v16bf r;
#pragma unroll
  for (int i = 0; i < 8; ++i) { r[i] = lo[i]; r[i + 8] = hi[i]; }
  return r;
}

// A fragment (16 rows x 32 K, bf16), A row-major with leading dim ld.
// lane L holds row (row0 + L&15); elements 0..7 = K kbase..kbase+7,
// elements 8..15 = K kbase+16..kbase+23, kbase = (L<16 ? 0 : 8).
DEVINL v16bf load_frag_a(const __bf16* A, int ld, int row0, int k0, int lane) {
  int r  = row0 + (lane & 15);
  int kb = k0 + ((lane < 16) ? 0 : 8);
  const __bf16* p = A + (size_t)r * ld + kb;
  return frag_cat(*(const v8bf*)p, *(const v8bf*)(p + 16));
}

// B fragment (32 K x 16 cols) from a transposed (col-major-as-rows) matrix
// Wt[n][k] with leading dim ld: lane L = column (n0 + L&15),
// elements 0..7 = K kbase..kbase+7, elements 8..15 = K kbase+16..kbase+23, kbase = k0 + (L<16?0:8).
DEVINL v16bf load_frag_bt(const __bf16* Wt, int ld, int n0, int k0, int lane) {
  int n  = n0 + (lane & 15);
  int kb = k0 + ((lane < 16) ? 0 : 8);
  const __bf16* p = Wt + (size_t)n * ld + kb;
  return frag_cat(*(const v8bf*)p, *(const v8bf*)(p + 16));
}

DEVINL v8f wmma_bf16(v16bf a, v16bf b, v8f c) {
  return __builtin_amdgcn_wmma_f32_16x16x32_bf16(false, a, false, b, (short)0, c,
                                                 false, false);
}

// Async global->LDS copy of 16 bytes per lane (GLOBAL_LOAD_ASYNC_TO_LDS_B128,
// tracked by ASYNCcnt). VDST VGPR carries the wave-relative LDS byte address,
// which per the ISA aperture mapping equals addr[31:0] of the generic pointer.
DEVINL void async_copy_b128(const void* gsrc, void* lds_dst) {
  asm volatile("global_load_async_to_lds_b128 %0, %1, off"
               :
               : "v"((unsigned)(uintptr_t)lds_dst),
                 "v"((unsigned long long)(uintptr_t)gsrc)
               : "memory");
}
DEVINL void wait_async0() { asm volatile("s_wait_asynccnt 0" ::: "memory"); }

// ---------------- constants ----------------
#define D_MODEL 1024
#define D_INNER 2048
#define NHEADS  32
#define HEADDIM 64
#define NBLK    32
#define CHUNK   64
#define SEQ     2048
#define BATCH   2
#define ROWS    (BATCH * SEQ)          // 4096
#define CONV_DIM 2176
#define D_IN_PROJ 4256
#define LORA_K  64                     // padded from 48

// ---------------- conversions ----------------
__global__ void k_f32_to_bf16(const float* __restrict__ src, __bf16* __restrict__ dst, int n) {
  int i = blockIdx.x * 256 + threadIdx.x;
  if (i < n) dst[i] = (__bf16)src[i];
}

// theta_w1 (1024,48) -> w1t (48,1024)
__global__ void k_transpose_w1(const float* __restrict__ w1, __bf16* __restrict__ w1t) {
  int i = blockIdx.x * 256 + threadIdx.x;   // 48*1024
  if (i >= 48 * 1024) return;
  int n = i >> 10, k = i & 1023;
  w1t[i] = (__bf16)w1[k * 48 + n];
}

// theta_w2 (48,1024) -> w2t (1024,64) zero padded in k
__global__ void k_transpose_w2(const float* __restrict__ w2, __bf16* __restrict__ w2t) {
  int i = blockIdx.x * 256 + threadIdx.x;   // 1024*64
  if (i >= 1024 * 64) return;
  int n = i >> 6, k = i & 63;
  w2t[i] = (k < 48) ? (__bf16)w2[k * 1024 + n] : (__bf16)0.0f;
}

// ---------------- generic GEMM: C[M,N] = A[M,K] @ W[N,K]^T ----------------
// 256 threads = 8 waves; wave tile 64x32 (4x2 WMMA tiles), block tile 256x64.
// N-edge handled by clamping load columns (N % 32 == 0) and guarding stores,
// keeping the K-loop branch-free (no per-iteration exec masking).
__global__ void k_gemm_bf16_nt(const __bf16* __restrict__ A, const __bf16* __restrict__ W,
                               float* __restrict__ C, int M, int N, int K) {
  int lane = threadIdx.x & 31, wid = threadIdx.x >> 5;
  int m0 = blockIdx.x * 256 + (wid >> 1) * 64;
  int n0 = blockIdx.y * 64 + (wid & 1) * 32;
  int n0c = (n0 <= N - 32) ? n0 : (N - 32);     // clamped base for loads
  v8f acc[4][2];
#pragma unroll
  for (int i = 0; i < 4; ++i) { acc[i][0] = {}; acc[i][1] = {}; }

  for (int k0 = 0; k0 < K; k0 += 32) {
    if (k0 + 64 < K) {   // stream-ahead hint (global_prefetch_b8)
      __builtin_prefetch(A + (size_t)(m0 + (lane & 15)) * K + k0 + 64, 0, 3);
      __builtin_prefetch(W + (size_t)(n0c + (lane & 15)) * K + k0 + 64, 0, 3);
    }
    v16bf b0 = load_frag_bt(W, K, n0c, k0, lane);
    v16bf b1 = load_frag_bt(W, K, n0c + 16, k0, lane);
#pragma unroll
    for (int i = 0; i < 4; ++i) {
      v16bf a = load_frag_a(A, K, m0 + 16 * i, k0, lane);
      acc[i][0] = wmma_bf16(a, b0, acc[i][0]);
      acc[i][1] = wmma_bf16(a, b1, acc[i][1]);
    }
  }

  bool ok0 = n0 < N, ok1 = (n0 + 16) < N;
  int cn = lane & 15;
  int rb = (lane < 16) ? 0 : 8;
#pragma unroll
  for (int i = 0; i < 4; ++i) {
#pragma unroll
    for (int v = 0; v < 8; ++v) {
      int r = m0 + 16 * i + rb + v;
      if (ok0) C[(size_t)r * N + n0 + cn]      = acc[i][0][v];
      if (ok1) C[(size_t)r * N + n0 + 16 + cn] = acc[i][1][v];
    }
  }
}

// ---------------- dt = softplus(zx[...,-32:] + dt_bias) ----------------
__global__ void k_dt(const float* __restrict__ zx, const float* __restrict__ dt_bias,
                     float* __restrict__ dtb) {
  int i = blockIdx.x * 256 + threadIdx.x;       // ROWS*NHEADS
  if (i >= ROWS * NHEADS) return;
  int m = i >> 5, h = i & 31;
  float v = zx[(size_t)m * D_IN_PROJ + (D_INNER + CONV_DIM) + h] + dt_bias[h];
  dtb[i] = (v > 20.f) ? v : log1pf(__expf(v));
}

// ---------------- depthwise causal conv (k=4) + SiLU ----------------
__global__ void k_conv(const float* __restrict__ zx, const float* __restrict__ cw,
                       const float* __restrict__ cb, float* __restrict__ xc) {
  long i = (long)blockIdx.x * 256 + threadIdx.x;  // ROWS*CONV_DIM
  if (i >= (long)ROWS * CONV_DIM) return;
  int c = (int)(i % CONV_DIM);
  long ml = i / CONV_DIM;                 // b*SEQ + l
  int l = (int)(ml & (SEQ - 1));
  float acc = cb[c];
#pragma unroll
  for (int k = 0; k < 4; ++k) {
    int ls = l - 3 + k;
    float xv = (ls >= 0) ? zx[(ml - 3 + k) * D_IN_PROJ + D_INNER + c] : 0.f;
    acc = fmaf(xv, cw[c * 4 + k], acc);
  }
  xc[i] = acc / (1.f + __expf(-acc));     // SiLU
}

// ---------------- lora stage 1: t1 = tanh(x @ w1), K-padded to 64 ----------
__global__ void k_lora1(const __bf16* __restrict__ xb, const __bf16* __restrict__ w1t,
                        __bf16* __restrict__ t1) {
  int lane = threadIdx.x & 31, wid = threadIdx.x >> 5;   // 128 threads, 4 waves
  int m0 = blockIdx.x * 64 + wid * 16;
  v8f c0 = {}, c1 = {}, c2 = {};
  for (int k0 = 0; k0 < D_MODEL; k0 += 32) {
    v16bf a = load_frag_a(xb, D_MODEL, m0, k0, lane);
    c0 = wmma_bf16(a, load_frag_bt(w1t, D_MODEL, 0, k0, lane), c0);
    c1 = wmma_bf16(a, load_frag_bt(w1t, D_MODEL, 16, k0, lane), c1);
    c2 = wmma_bf16(a, load_frag_bt(w1t, D_MODEL, 32, k0, lane), c2);
  }
  int cn = lane & 15;
  int rb = m0 + ((lane < 16) ? 0 : 8);
#pragma unroll
  for (int v = 0; v < 8; ++v) {
    size_t r = (size_t)(rb + v) * LORA_K;
    t1[r + cn]      = (__bf16)tanhf(c0[v]);
    t1[r + 16 + cn] = (__bf16)tanhf(c1[v]);
    t1[r + 32 + cn] = (__bf16)tanhf(c2[v]);
    t1[r + 48 + cn] = (__bf16)0.0f;       // K padding
  }
}

// ---------------- lora stage 2 + theta/lambda -> log_z ----------------
DEVINL void lz_epilogue(v8f acc, int row0, int col0, int lane,
                        const float* __restrict__ dtb, const float* __restrict__ tb,
                        const float* __restrict__ lb, const float* __restrict__ eta,
                        float* __restrict__ lzr, float* __restrict__ lzi) {
  int n = col0 + (lane & 15);
  int h = n >> 5;
  float th0 = tb[n], lam0 = lb[n], et = eta[n];
  int rb = row0 + ((lane < 16) ? 0 : 8);
#pragma unroll
  for (int v = 0; v < 8; ++v) {
    int m = rb + v;
    float th = th0 + acc[v];
    th = fminf(fmaxf(th, -1.5707963267948966f), 1.5707963267948966f);
    float lam = fmaf(et, th * th, lam0);
    float dtv = dtb[m * NHEADS + h];
    lzr[(size_t)m * 1024 + n] = -lam * dtv;
    lzi[(size_t)m * 1024 + n] = th * dtv;
  }
}

__global__ void k_lora2(const __bf16* __restrict__ t1, const __bf16* __restrict__ w2t,
                        const float* __restrict__ dtb, const float* __restrict__ tb,
                        const float* __restrict__ lb, const float* __restrict__ eta,
                        float* __restrict__ lzr, float* __restrict__ lzi) {
  int lane = threadIdx.x & 31, wid = threadIdx.x >> 5;
  int m0 = blockIdx.x * 128 + (wid >> 1) * 32;
  int n0 = blockIdx.y * 64 + (wid & 1) * 32;
  v8f c00 = {}, c01 = {}, c10 = {}, c11 = {};
  for (int k0 = 0; k0 < LORA_K; k0 += 32) {
    v16bf a0 = load_frag_a(t1, LORA_K, m0, k0, lane);
    v16bf a1 = load_frag_a(t1, LORA_K, m0 + 16, k0, lane);
    v16bf b0 = load_frag_bt(w2t, LORA_K, n0, k0, lane);
    v16bf b1 = load_frag_bt(w2t, LORA_K, n0 + 16, k0, lane);
    c00 = wmma_bf16(a0, b0, c00); c01 = wmma_bf16(a0, b1, c01);
    c10 = wmma_bf16(a1, b0, c10); c11 = wmma_bf16(a1, b1, c11);
  }
  lz_epilogue(c00, m0,      n0,      lane, dtb, tb, lb, eta, lzr, lzi);
  lz_epilogue(c01, m0,      n0 + 16, lane, dtb, tb, lb, eta, lzr, lzi);
  lz_epilogue(c10, m0 + 16, n0,      lane, dtb, tb, lb, eta, lzr, lzi);
  lz_epilogue(c11, m0 + 16, n0 + 16, lane, dtb, tb, lb, eta, lzr, lzi);
}

// ---------------- chunked complex scan (one WG per (b,h)) ----------------
__global__ void k_scan(const float* __restrict__ xc,   // (ROWS, CONV_DIM) conv output
                       const float* __restrict__ lzr, const float* __restrict__ lzi,
                       const float* __restrict__ dtb,  // (ROWS, NHEADS)
                       const float* __restrict__ Dv,   // (NHEADS)
                       float* __restrict__ yb) {       // (ROWS, D_INNER)
  int b = blockIdx.x >> 5;
  int h = blockIdx.x & 31;
  int tid = threadIdx.x;
  int lane = tid & 31, wid = tid >> 5;

  __shared__ __align__(32) float  s_cum[2][64][32];   // cumlog (G overlays [0])
  __shared__ __align__(32) __bf16 s_P[2][64][32];     // Pr, -Pi  (t,k)
  __shared__ __align__(32) __bf16 s_Q[2][64][32];     // Qr, Qi   (s,k)
  __shared__ __align__(32) __bf16 s_XdT[64][64];      // x*dt     (p,s)
  __shared__ __align__(32) __bf16 s_Qp[2][32][64];    // Q' for state (k,s)
  __shared__ __align__(32) __bf16 s_Sb[2][64][32];    // exp(m_k)-scaled state (p,k)
  __shared__ float s_mk[32], s_em[32], s_dr[32], s_di[32], s_c63r[32], s_c63i[32];
  __shared__ float s_dt[64];
  __bf16* s_G = (__bf16*)&s_cum[0][0][0];             // 64x64 masked gate matrix

  const float Dh = Dv[h];
  // recurrent state in registers, mapped to the phase-5 C-fragment layout
  float Sr[8], Si[8];
#pragma unroll
  for (int v = 0; v < 8; ++v) { Sr[v] = 0.f; Si[v] = 0.f; }
  const int pp = (wid >> 1) * 16;   // this wave's p-tile
  const int kk = (wid & 1) * 16;    // this wave's k-tile

  for (int c = 0; c < SEQ / CHUNK; ++c) {
    const size_t rowbase = (size_t)b * SEQ + (size_t)c * CHUNK;

    // ---- phase 0: async DMA of this chunk's log_z into LDS (ASYNCcnt path)
    for (int idx = tid; idx < 512; idx += 256) {
      int t = idx >> 3, k4 = (idx & 7) * 4;             // 4 floats = 16 B per op
      size_t g = (rowbase + t) * 1024 + (size_t)h * 32 + k4;
      async_copy_b128(lzr + g, &s_cum[0][t][k4]);
      async_copy_b128(lzi + g, &s_cum[1][t][k4]);
    }
    if (tid >= 64 && tid < 128) {
      int t = tid - 64;
      s_dt[t] = dtb[(rowbase + t) * NHEADS + h];
    }
    wait_async0();
    __syncthreads();

    // ---- phase 1: in-LDS cumulative sums + per-k chunk constants
    if (tid < 32) {
      int k = tid;
      float ar = 0.f, ai = 0.f;
      for (int t = 0; t < 64; ++t) {
        ar += s_cum[0][t][k]; ai += s_cum[1][t][k];
        s_cum[0][t][k] = ar; s_cum[1][t][k] = ai;
        if (t == 31) { s_mk[k] = ar; s_em[k] = __expf(ar); }
      }
      s_c63r[k] = ar; s_c63i[k] = ai;
      float er = __expf(ar), sn, cs;
      __sincosf(ai, &sn, &cs);
      s_dr[k] = er * cs; s_di[k] = er * sn;       // chunk decay
    }
    __syncthreads();

    // ---- phase 2: build bf16 operand panels
    for (int idx = tid; idx < 2048; idx += 256) {
      int t = idx >> 5, k = idx & 31;
      float clr = s_cum[0][t][k], cli = s_cum[1][t][k], mk = s_mk[k];
      size_t roff = (rowbase + t) * CONV_DIM;
      float ccr = xc[roff + 2112 + 2 * k], cci = xc[roff + 2112 + 2 * k + 1];
      float bcr = xc[roff + 2048 + 2 * k], bci = xc[roff + 2048 + 2 * k + 1];
      float sn, cs; __sincosf(cli, &sn, &cs);
      float ep = __expf(clr - mk);
      s_P[0][t][k] = (__bf16)( ep * (ccr * cs - cci * sn));        //  Re(cc e^{cl-m})
      s_P[1][t][k] = (__bf16)(-ep * (ccr * sn + cci * cs));        // -Im
      float eq = __expf(mk - clr);
      s_Q[0][t][k] = (__bf16)( eq * (bcr * cs + bci * sn));        //  Re(bc e^{m-cl})
      s_Q[1][t][k] = (__bf16)( eq * (bci * cs - bcr * sn));        //  Im
      float eqp = __expf(s_c63r[k] - clr);
      float dA = s_c63i[k] - cli, sn2, cs2;
      __sincosf(dA, &sn2, &cs2);
      s_Qp[0][k][t] = (__bf16)(eqp * (bcr * cs2 - bci * sn2));     // Q' = bc e^{cl63-cl}
      s_Qp[1][k][t] = (__bf16)(eqp * (bcr * sn2 + bci * cs2));
    }
    for (int idx = tid; idx < 4096; idx += 256) {
      int p = idx & 63, s = idx >> 6;
      float xv = xc[(rowbase + s) * CONV_DIM + (size_t)h * HEADDIM + p];
      s_XdT[p][s] = (__bf16)(xv * s_dt[s]);
    }
#pragma unroll
    for (int v = 0; v < 8; ++v) {                // scaled state -> bf16 panels
      int p = pp + v + ((lane < 16) ? 0 : 8);
      int k = kk + (lane & 15);
      float em = s_em[k];
      s_Sb[0][p][k] = (__bf16)(Sr[v] * em);
      s_Sb[1][p][k] = (__bf16)(Si[v] * em);
    }
    __syncthreads();

    // ---- phase 3: G = Re(P Q^T) = Pr@Qr^T + (-Pi)@Qi^T, causal mask, -> LDS
#pragma unroll
    for (int r = 0; r < 2; ++r) {
      int id = wid + r * 8;
      int tt = (id >> 2) * 16, ts = (id & 3) * 16;
      v8f acc = {};
      acc = wmma_bf16(load_frag_a(&s_P[0][0][0], 32, tt, 0, lane),
                      load_frag_bt(&s_Q[0][0][0], 32, ts, 0, lane), acc);
      acc = wmma_bf16(load_frag_a(&s_P[1][0][0], 32, tt, 0, lane),
                      load_frag_bt(&s_Q[1][0][0], 32, ts, 0, lane), acc);
      int sc = ts + (lane & 15);
      int rb = tt + ((lane < 16) ? 0 : 8);
#pragma unroll
      for (int v = 0; v < 8; ++v) {
        int t = rb + v;
        s_G[t * 64 + sc] = (__bf16)((sc <= t) ? acc[v] : 0.f);   // mask kills inf too
      }
    }
    __syncthreads();

    // ---- phase 4: Y = G@Xd + Pr@SbR + (-Pi)@SbI ; add D-skip, store y
#pragma unroll
    for (int r = 0; r < 2; ++r) {
      int id = wid + r * 8;
      int tt = (id >> 2) * 16, pc0 = (id & 3) * 16;
      v8f acc = {};
      acc = wmma_bf16(load_frag_a(s_G, 64, tt, 0, lane),
                      load_frag_bt(&s_XdT[0][0], 64, pc0, 0, lane), acc);
      acc = wmma_bf16(load_frag_a(s_G, 64, tt, 32, lane),
                      load_frag_bt(&s_XdT[0][0], 64, pc0, 32, lane), acc);
      acc = wmma_bf16(load_frag_a(&s_P[0][0][0], 32, tt, 0, lane),
                      load_frag_bt(&s_Sb[0][0][0], 32, pc0, 0, lane), acc);
      acc = wmma_bf16(load_frag_a(&s_P[1][0][0], 32, tt, 0, lane),
                      load_frag_bt(&s_Sb[1][0][0], 32, pc0, 0, lane), acc);
      int p = pc0 + (lane & 15);
      int rb = tt + ((lane < 16) ? 0 : 8);
#pragma unroll
      for (int v = 0; v < 8; ++v) {
        size_t row = rowbase + rb + v;
        float xs = xc[row * CONV_DIM + (size_t)h * HEADDIM + p];
        yb[row * D_INNER + (size_t)h * HEADDIM + p] = acc[v] + xs * Dh;
      }
    }

    // ---- phase 5: state' = decay*state + Xd^T @ Q'^T (complex)
    {
      v8f zr = {}, zi = {};
      v16bf a0 = load_frag_a(&s_XdT[0][0], 64, pp, 0, lane);
      v16bf a1 = load_frag_a(&s_XdT[0][0], 64, pp, 32, lane);
      zr = wmma_bf16(a0, load_frag_bt(&s_Qp[0][0][0], 64, kk, 0, lane), zr);
      zr = wmma_bf16(a1, load_frag_bt(&s_Qp[0][0][0], 64, kk, 32, lane), zr);
      zi = wmma_bf16(a0, load_frag_bt(&s_Qp[1][0][0], 64, kk, 0, lane), zi);
      zi = wmma_bf16(a1, load_frag_bt(&s_Qp[1][0][0], 64, kk, 32, lane), zi);
      int k = kk + (lane & 15);
      float dr = s_dr[k], di = s_di[k];
#pragma unroll
      for (int v = 0; v < 8; ++v) {
        float nr = dr * Sr[v] - di * Si[v] + zr[v];
        float ni = dr * Si[v] + di * Sr[v] + zi[v];
        Sr[v] = nr; Si[v] = ni;
      }
    }
    __syncthreads();
  }
}

// ---------------- gating + RMSNorm -> bf16 ----------------
__global__ void k_gate_rms(const float* __restrict__ zx, const float* __restrict__ yb,
                           const float* __restrict__ nw, __bf16* __restrict__ gb) {
  int m = blockIdx.x, tid = threadIdx.x;
  float g[8], ss = 0.f;
#pragma unroll
  for (int u = 0; u < 8; ++u) {
    int i = tid + u * 256;
    float z = zx[(size_t)m * D_IN_PROJ + i];
    float y = yb[(size_t)m * D_INNER + i];
    float v = y * (z / (1.f + __expf(-z)));
    g[u] = v; ss += v * v;
  }
  __shared__ float red[256];
  red[tid] = ss; __syncthreads();
  for (int off = 128; off > 0; off >>= 1) {
    if (tid < off) red[tid] += red[tid + off];
    __syncthreads();
  }
  float scale = rsqrtf(red[0] * (1.f / D_INNER) + 1e-5f);
#pragma unroll
  for (int u = 0; u < 8; ++u) {
    int i = tid + u * 256;
    gb[(size_t)m * D_INNER + i] = (__bf16)(g[u] * scale * nw[i]);
  }
}

// ---------------- host launch ----------------
extern "C" void kernel_launch(void* const* d_in, const int* in_sizes, int n_in,
                              void* d_out, int out_size, void* d_ws, size_t ws_size,
                              hipStream_t stream) {
  (void)in_sizes; (void)n_in; (void)out_size; (void)ws_size;
  const float* x          = (const float*)d_in[0];
  const float* in_proj_w  = (const float*)d_in[1];
  const float* conv_w     = (const float*)d_in[2];
  const float* conv_b     = (const float*)d_in[3];
  const float* dt_bias    = (const float*)d_in[4];
  const float* lambda_b   = (const float*)d_in[5];
  const float* theta_b    = (const float*)d_in[6];
  const float* theta_w1   = (const float*)d_in[7];
  const float* theta_w2   = (const float*)d_in[8];
  const float* visc_eta   = (const float*)d_in[9];
  const float* Dv         = (const float*)d_in[10];
  const float* norm_w     = (const float*)d_in[11];
  const float* out_proj_w = (const float*)d_in[12];

  char* ws = (char*)d_ws;
  size_t off = 0;
  auto take = [&](size_t bytes) { char* p = ws + off; off += (bytes + 255) & ~size_t(255); return p; };
  __bf16* xb   = (__bf16*)take((size_t)ROWS * D_MODEL * 2);        // x bf16
  __bf16* wib  = (__bf16*)take((size_t)D_IN_PROJ * D_MODEL * 2);   // in_proj bf16
  __bf16* wob  = (__bf16*)take((size_t)D_MODEL * D_INNER * 2);     // out_proj bf16
  __bf16* w1t  = (__bf16*)take((size_t)48 * D_MODEL * 2);
  __bf16* w2t  = (__bf16*)take((size_t)1024 * LORA_K * 2);
  float*  zx   = (float*)take((size_t)ROWS * D_IN_PROJ * 4);       // zxbcdt
  float*  dtb  = (float*)take((size_t)ROWS * NHEADS * 4);
  float*  xcv  = (float*)take((size_t)ROWS * CONV_DIM * 4);        // conv+SiLU
  __bf16* t1   = (__bf16*)take((size_t)ROWS * LORA_K * 2);
  float*  lzr  = (float*)take((size_t)ROWS * 1024 * 4);
  float*  lzi  = (float*)take((size_t)ROWS * 1024 * 4);
  float*  ybuf = (float*)take((size_t)ROWS * D_INNER * 4);
  __bf16* gb   = (__bf16*)take((size_t)ROWS * D_INNER * 2);

  // 0) conversions
  k_f32_to_bf16<<<(ROWS * D_MODEL) / 256, 256, 0, stream>>>(x, xb, ROWS * D_MODEL);
  k_f32_to_bf16<<<(D_IN_PROJ * D_MODEL) / 256, 256, 0, stream>>>(in_proj_w, wib, D_IN_PROJ * D_MODEL);
  k_f32_to_bf16<<<(D_MODEL * D_INNER) / 256, 256, 0, stream>>>(out_proj_w, wob, D_MODEL * D_INNER);
  k_transpose_w1<<<(48 * 1024) / 256, 256, 0, stream>>>(theta_w1, w1t);
  k_transpose_w2<<<(1024 * 64) / 256, 256, 0, stream>>>(theta_w2, w2t);

  // 1) in_proj GEMM: (4096 x 1024) @ (1024 x 4256)
  k_gemm_bf16_nt<<<dim3(ROWS / 256, (D_IN_PROJ + 63) / 64), 256, 0, stream>>>(
      xb, wib, zx, ROWS, D_IN_PROJ, D_MODEL);

  // 2) dt + conv
  k_dt<<<(ROWS * NHEADS) / 256, 256, 0, stream>>>(zx, dt_bias, dtb);
  k_conv<<<(ROWS * CONV_DIM) / 256, 256, 0, stream>>>(zx, conv_w, conv_b, xcv);

  // 3) lora -> log_z
  k_lora1<<<ROWS / 64, 128, 0, stream>>>(xb, w1t, t1);
  k_lora2<<<dim3(ROWS / 128, 1024 / 64), 256, 0, stream>>>(
      t1, w2t, dtb, theta_b, lambda_b, visc_eta, lzr, lzi);

  // 4) chunked scan
  k_scan<<<BATCH * NHEADS, 256, 0, stream>>>(xcv, lzr, lzi, dtb, Dv, ybuf);

  // 5) gate + RMSNorm
  k_gate_rms<<<ROWS, 256, 0, stream>>>(zx, ybuf, norm_w, gb);

  // 6) out_proj GEMM -> d_out
  k_gemm_bf16_nt<<<dim3(ROWS / 256, D_MODEL / 64), 256, 0, stream>>>(
      gb, wob, (float*)d_out, ROWS, D_MODEL, D_INNER);
}
